// SimpleRNN_32573031973131
// MI455X (gfx1250) — matrix operand order repaired and checked by reading, not compiled, not run on hardware
//
#include <hip/hip_runtime.h>

typedef __attribute__((ext_vector_type(16))) __bf16 v16bf;
typedef __attribute__((ext_vector_type(8)))  __bf16 v8bf;
typedef __attribute__((ext_vector_type(8)))  float  v8f;
typedef int v4i_vs __attribute__((vector_size(16)));   // matches builtin param type

#define AS_GLOBAL __attribute__((address_space(1)))
#define AS_SHARED __attribute__((address_space(3)))

#define BT    16384   // B*T
#define Hdim  1024
#define Kd    256
#define Vd    256
#define Od    1024
#define Tlen  2048
#define Bb    8

#if __has_builtin(__builtin_amdgcn_global_load_async_to_lds_b128) && \
    __has_builtin(__builtin_amdgcn_s_wait_asynccnt)
#define HAS_ASYNC_LDS 1
#else
#define HAS_ASYNC_LDS 0
#endif

// ---------------------------------------------------------------------------
// Kernel 0a: f32 -> bf16 convert of hidden_state (BT x H)
// ---------------------------------------------------------------------------
__global__ void cvt_hidden(const float* __restrict__ X, __bf16* __restrict__ X16) {
    size_t i0 = ((size_t)blockIdx.x * blockDim.x + threadIdx.x) * 4;
    float4 f = *(const float4*)(X + i0);
    __bf16* p = X16 + i0;
    p[0] = (__bf16)f.x; p[1] = (__bf16)f.y; p[2] = (__bf16)f.z; p[3] = (__bf16)f.w;
}

// ---------------------------------------------------------------------------
// Kernel 0b: build transposed bf16 weights.
//   WtAll[n][h], n in [0,1024): [Wq|Wk|Wg|Wv] columns; W* are (H x 256) row-major
//   WoT[o][v]  = Wo[v][o], Wo is (256 x 1024) row-major
// ---------------------------------------------------------------------------
__global__ void build_wt(const float* __restrict__ Wq, const float* __restrict__ Wk,
                         const float* __restrict__ Wg, const float* __restrict__ Wv,
                         const float* __restrict__ Wo,
                         __bf16* __restrict__ WtAll, __bf16* __restrict__ WoT) {
    int idx = blockIdx.x * blockDim.x + threadIdx.x;   // 0 .. 1024*1024-1
    int n = idx >> 10, h = idx & 1023;
    float w;
    if (n < 256)      w = Wq[h * 256 + n];
    else if (n < 512) w = Wk[h * 256 + (n - 256)];
    else if (n < 768) w = Wg[h * 256 + (n - 512)];
    else              w = Wv[h * 256 + (n - 768)];
    WtAll[idx] = (__bf16)w;
    if (idx < 1024 * 256) {
        int o = idx >> 8, v = idx & 255;
        WoT[idx] = (__bf16)Wo[v * 1024 + o];
    }
}

// ---------------------------------------------------------------------------
// WMMA bf16 GEMM:  Y[m][n] = epilogue( sum_k A[m][k] * Bt[n][k] )
//   A:  (BT x KDIM) bf16 row-major
//   Bt: (1024 x KDIM) bf16 row-major (pre-transposed weights; L2-resident)
//   Wave computes a 16(m) x 32(n) tile (2 accumulators, A-fragment reuse).
//   Workgroup = 8 waves arranged 4(m) x 2(n) -> 64 x 64 tile.
// MODE 0: qkgv epilogue (bias + sigmoid on k/g columns), MODE 1: + bo only.
// ---------------------------------------------------------------------------
template<int MODE>
__device__ __forceinline__ float epilogue(float x, int col,
                                          const float* b0, const float* b1,
                                          const float* b2, const float* b3) {
    if (MODE == 0) {
        if (col < 256) return x + b0[col];
        if (col < 512) { x += b1[col - 256]; return 1.f / (1.f + __expf(-x)); }
        if (col < 768) { x += b2[col - 512]; return 1.f / (1.f + __expf(-x)); }
        return x + b3[col - 768];
    } else {
        return x + b0[col];
    }
}

template<int KDIM, int MODE>
__global__ void wmma_gemm(const __bf16* __restrict__ A, const __bf16* __restrict__ Bt,
                          const float* __restrict__ b0, const float* __restrict__ b1,
                          const float* __restrict__ b2, const float* __restrict__ b3,
                          float* __restrict__ Y) {
    const int N = 1024;
    int lane = threadIdx.x & 31;
    int wave = threadIdx.x >> 5;          // 0..7
    int wm = wave >> 1;                   // 0..3
    int wn = wave & 1;                    // 0..1
    int m0 = blockIdx.x * 64 + wm * 16;
    int n0 = blockIdx.y * 64 + wn * 32;

    bool hi = lane >= 16;
    int  l15 = lane & 15;

    const __bf16* xrow  = A  + (size_t)(m0 + l15) * KDIM;
    const __bf16* w0row = Bt + (size_t)(n0 + l15) * KDIM;
    const __bf16* w1row = Bt + (size_t)(n0 + 16 + l15) * KDIM;

    v8f acc0 = {}; v8f acc1 = {};

    for (int k0 = 0; k0 < KDIM; k0 += 32) {
        // A fragment (16x32 bf16): lanes 0-15 hold K {k0..+15},
        // lanes 16-31 hold K {k0+16..+31}
        v8bf a_lo = *(const v8bf*)(xrow + k0 + (hi ? 16 : 0));
        v8bf a_hi = *(const v8bf*)(xrow + k0 + 8 + (hi ? 16 : 0));
        v16bf a;
#pragma unroll
        for (int i = 0; i < 8; ++i) { a[i] = a_lo[i]; a[8 + i] = a_hi[i]; }
        // B fragment (32x16 bf16): lane n, 16 contiguous K values
        v16bf bf0 = *(const v16bf*)(w0row + k0 + (hi ? 16 : 0));
        v16bf bf1 = *(const v16bf*)(w1row + k0 + (hi ? 16 : 0));
        acc0 = __builtin_amdgcn_wmma_f32_16x16x32_bf16(false, a, false, bf0,
                                                       (short)0, acc0, false, false);
        acc1 = __builtin_amdgcn_wmma_f32_16x16x32_bf16(false, a, false, bf1,
                                                       (short)0, acc1, false, false);
    }

    int col0 = n0 + l15;
    int col1 = n0 + 16 + l15;
    int rbase = m0 + (hi ? 8 : 0);
#pragma unroll
    for (int r = 0; r < 8; ++r) {
        int row = rbase + r;
        Y[(size_t)row * N + col0] = epilogue<MODE>(acc0[r], col0, b0, b1, b2, b3);
        Y[(size_t)row * N + col1] = epilogue<MODE>(acc1[r], col1, b0, b1, b2, b3);
    }
}

// ---------------------------------------------------------------------------
// GLA scan: state[b,k,v] = state*g_t[k] + k_t[k]*v_t[v]; o_t[v] = sum_k q_t[k]*state
//   Y layout per (b,t) row of 1024 f32: [q | k_sig | g_sig | v]
//   Grid: 32 blocks = 8 batches x 4 V-quarters. Block = 256 threads.
//   Thread: col = quarter*64 + (tid>>2), k-slice = (tid&3)*64; 64 f32 state regs.
//   Double-buffered GLOBAL_LOAD_ASYNC_TO_LDS_B128 prefetch of row t+1 hides the
//   HBM/L2 latency that would otherwise sit inside the serial T-dependency chain.
//   Writes o as bf16 (A-matrix of the output GEMM) and final state f32 to d_out.
// ---------------------------------------------------------------------------
__global__ void gla_scan(const float* __restrict__ Y, __bf16* __restrict__ O16,
                         float* __restrict__ stateOut) {
    __shared__ float lds[2048];            // two 4KB row buffers
    int b       = blockIdx.x >> 2;
    int quarter = blockIdx.x & 3;
    int tid = threadIdx.x;                 // 0..255
    int col = quarter * 64 + (tid >> 2);
    int ks  = tid & 3;
    int k0  = ks * 64;

    float s[64];
#pragma unroll
    for (int j = 0; j < 64; ++j) s[j] = 0.f;

    const float* yb = Y + (size_t)b * Tlen * 1024;

    // per-thread 16B slice of one [q|k|g|v] row -> LDS buffer `buf`
    auto issue_row = [&](int t, int buf) {
        const float* src = yb + (size_t)t * 1024 + tid * 4;
        float* dst = lds + buf * 1024 + tid * 4;
#if HAS_ASYNC_LDS
        __builtin_amdgcn_global_load_async_to_lds_b128(
            (AS_GLOBAL v4i_vs*)src, (AS_SHARED v4i_vs*)dst, 0, 0);
#else
        *(float4*)dst = *(const float4*)src;
#endif
    };

    issue_row(0, 0);                       // prime buffer 0 with row t=0

    for (int t = 0; t < Tlen; ++t) {
        if (t + 1 < Tlen) {
            issue_row(t + 1, (t + 1) & 1); // prefetch next row into alt buffer
#if HAS_ASYNC_LDS
            __builtin_amdgcn_s_wait_asynccnt(1);  // retire oldest (row t)
#endif
        } else {
#if HAS_ASYNC_LDS
            __builtin_amdgcn_s_wait_asynccnt(0);  // last row: drain fully
#endif
        }
        __syncthreads();                   // row t visible to all waves

        const float* buf = lds + (t & 1) * 1024;
        float vt = buf[768 + col];
        const float* q  = buf + k0;
        const float* kk = buf + 256 + k0;
        const float* g  = buf + 512 + k0;
        float acc = 0.f;
#pragma unroll
        for (int j = 0; j < 64; ++j) {
            float sv = fmaf(s[j], g[j], kk[j] * vt);
            s[j] = sv;
            acc = fmaf(q[j], sv, acc);
        }
        // reduce the 4 k-slices (lanes 4c..4c+3 of the same wave)
        acc += __shfl_xor(acc, 1, 4);
        acc += __shfl_xor(acc, 2, 4);
        if (ks == 0) O16[((size_t)b * Tlen + t) * Vd + col] = (__bf16)acc;
        __syncthreads();                   // all reads done before buf is reused
    }

    // final_state (B,K,V) -> second output of the reference tuple
#pragma unroll
    for (int j = 0; j < 64; ++j)
        stateOut[((size_t)b * Kd + k0 + j) * Vd + col] = s[j];
}

// ---------------------------------------------------------------------------
extern "C" void kernel_launch(void* const* d_in, const int* in_sizes, int n_in,
                              void* d_out, int out_size, void* d_ws, size_t ws_size,
                              hipStream_t stream) {
    const float* hidden = (const float*)d_in[0];
    const float* Wq = (const float*)d_in[1];
    const float* bq = (const float*)d_in[2];
    const float* Wk = (const float*)d_in[3];
    const float* bk = (const float*)d_in[4];
    const float* Wv = (const float*)d_in[5];
    const float* bv = (const float*)d_in[6];
    const float* Wg = (const float*)d_in[7];
    const float* bg = (const float*)d_in[8];
    const float* Wo = (const float*)d_in[9];
    const float* bo = (const float*)d_in[10];
    float* out = (float*)d_out;

    char* ws = (char*)d_ws;
    // X16 (32MB) is dead after proj GEMM; O16 (8MB) overlays it.
    __bf16* X16   = (__bf16*)(ws + 0);
    __bf16* O16   = (__bf16*)(ws + 0);
    __bf16* WtAll = (__bf16*)(ws + (size_t)33554432);
    __bf16* WoT   = (__bf16*)(ws + (size_t)35651584);
    float*  Ybuf  = (float*)(ws + (size_t)36175872);   // BT x 1024 f32 (64MB)

    float* stateOut = out + (size_t)BT * Od;

    cvt_hidden<<<BT * Hdim / (256 * 4), 256, 0, stream>>>(hidden, X16);
    build_wt<<<(1024 * 1024) / 256, 256, 0, stream>>>(Wq, Wk, Wg, Wv, Wo, WtAll, WoT);
    // fused q/k/g/v projection: (BT x 1024) = (BT x 1024)bf16 . (1024 x 1024)bf16
    wmma_gemm<Hdim, 0><<<dim3(BT / 64, 1024 / 64), 256, 0, stream>>>(
        X16, WtAll, bq, bk, bg, bv, Ybuf);
    // sequential gated scan over T, state in VGPRs, async LDS double buffering
    gla_scan<<<Bb * 4, 256, 0, stream>>>(Ybuf, O16, stateOut);
    // output projection: (BT x 1024) = (BT x 256)bf16 . (256 x 1024)bf16
    wmma_gemm<Kd, 1><<<dim3(BT / 64, 1024 / 64), 256, 0, stream>>>(
        O16, WoT, bo, bo, bo, bo, out);
}
